// Cross_MultiHead_Attention_26895085208063
// MI455X (gfx1250) — matrix operand order repaired and checked by reading, not compiled, not run on hardware
//
#include <hip/hip_runtime.h>
#include <hip/hip_bf16.h>

// Problem constants (reference: B=4, T=S=2048, C=1024, H=16, D=64)
#define BQ 4
#define TT 2048
#define SSZ 2048
#define CC 1024
#define HH 16
#define DD 64

typedef __attribute__((ext_vector_type(16))) __bf16 bf16x16;
typedef __attribute__((ext_vector_type(8)))  float  f32x8;

__device__ __forceinline__ f32x8 wmma_bf16(bf16x16 a, bf16x16 b, f32x8 c) {
  // D(16x16 f32) = A(16x32 bf16) * B(32x16 bf16) + C
  return __builtin_amdgcn_wmma_f32_16x16x32_bf16(false, a, false, b, (short)0, c,
                                                 false, false);
}

// A-matrix fragment (16x32, row-major in memory, leading dim `ld` elements).
// Lane l: row = l&15; VGPR0..3 hold K = kh*8 + {0..7}; VGPR4..7 hold K = 16+kh*8+{0..7}
// where kh = l>>4.  -> two 16B contiguous loads per lane.
__device__ __forceinline__ bf16x16 load_frag_a(const __bf16* p, int ld) {
  const int lane = threadIdx.x & 31;
  const __bf16* q = p + (size_t)(lane & 15) * ld + ((lane >> 4) << 3);
  bf16x16 f;
  ((uint4*)&f)[0] = *(const uint4*)(q);
  ((uint4*)&f)[1] = *(const uint4*)(q + 16);
  return f;
}

// B-matrix fragment (32x16 K-major). Memory holds B transposed: 16 rows (=N) x K,
// leading dim `ld`. Lane l: col n = l&15; VGPR0..3 hold K = kh*8 + {0..7}; VGPR4..7 hold
// K = 16+kh*8+{0..7} where kh = l>>4 -> two 16B loads per lane.
__device__ __forceinline__ bf16x16 load_frag_b(const __bf16* p, int ld) {
  const int lane = threadIdx.x & 31;
  const __bf16* q = p + (size_t)(lane & 15) * ld + ((lane >> 4) << 3);
  bf16x16 f;
  ((uint4*)&f)[0] = *(const uint4*)(q);
  ((uint4*)&f)[1] = *(const uint4*)(q + 16);
  return f;
}

// Prefetch one cacheline per lane covering a fragment's rows (global_prefetch_b8).
__device__ __forceinline__ void prefetch_frag(const __bf16* p, int ld) {
  const int lane = threadIdx.x & 31;
  __builtin_prefetch(p + (size_t)(lane & 15) * ld + ((lane >> 4) << 4), 0, 0);
}

// ---------------- conversion / layout kernels ----------------

__global__ __launch_bounds__(256) void cvt_f32_bf16_kernel(
    const float* __restrict__ in, __bf16* __restrict__ out, size_t n) {
  size_t i = (size_t)blockIdx.x * blockDim.x + threadIdx.x;
  if (i < n) out[i] = (__bf16)in[i];
}

// in: rows x cols (f32, row-major) -> out: cols x rows (bf16, row-major)
__global__ __launch_bounds__(256) void transpose_f32_bf16_kernel(
    const float* __restrict__ in, __bf16* __restrict__ out, int rows, int cols) {
  size_t i = (size_t)blockIdx.x * blockDim.x + threadIdx.x;
  if (i >= (size_t)rows * cols) return;
  const int r = (int)(i / cols);
  const int c = (int)(i % cols);
  out[(size_t)c * rows + r] = (__bf16)in[i];
}

// kv: (B*S, 2C) bf16; v part at column C + h*64 + d.  vT: (B,H,D,S) bf16.
__global__ __launch_bounds__(256) void build_vT_kernel(
    const __bf16* __restrict__ kv, __bf16* __restrict__ vT) {
  const size_t i = (size_t)blockIdx.x * blockDim.x + threadIdx.x; // 2^23 total
  const int s = (int)(i & (SSZ - 1));
  const int d = (int)((i >> 11) & (DD - 1));
  const int h = (int)((i >> 17) & (HH - 1));
  const int b = (int)(i >> 21);
  vT[i] = kv[((size_t)(b * SSZ + s)) * (2 * CC) + CC + h * DD + d];
}

// ---------------- bf16 WMMA GEMM: C(MxN) = A(MxK) * Bt(NxK)^T ----------------
// 256 threads = 8 waves (2x4), each wave computes a 32x64 tile -> 64x256 block tile.
// 8 WMMAs per 6 fragment loads per K-step.
template <typename OutT>
__global__ __launch_bounds__(256) void gemm_bf16_kernel(
    const __bf16* __restrict__ A, const __bf16* __restrict__ Bt,
    OutT* __restrict__ C, int M, int N, int K) {
  const int wid  = threadIdx.x >> 5;
  const int lane = threadIdx.x & 31;
  const int m0 = blockIdx.y * 64 + (wid >> 2) * 32;   // 2 m-tiles of 16
  const int n0 = blockIdx.x * 256 + (wid & 3) * 64;   // 4 n-tiles of 16

  const __bf16* Ab0 = A + (size_t)m0 * K;
  const __bf16* Ab1 = A + (size_t)(m0 + 16) * K;

  f32x8 acc[2][4] = {};
  for (int k0 = 0; k0 < K; k0 += 32) {
    if (k0 + 32 < K) {
      prefetch_frag(Ab0 + k0 + 32, K);
      prefetch_frag(Ab1 + k0 + 32, K);
#pragma unroll
      for (int j = 0; j < 4; ++j)
        prefetch_frag(Bt + (size_t)(n0 + 16 * j) * K + k0 + 32, K);
    }
    const bf16x16 a0 = load_frag_a(Ab0 + k0, K);
    const bf16x16 a1 = load_frag_a(Ab1 + k0, K);
#pragma unroll
    for (int j = 0; j < 4; ++j) {
      const bf16x16 b = load_frag_b(Bt + (size_t)(n0 + 16 * j) * K + k0, K);
      acc[0][j] = wmma_bf16(a0, b, acc[0][j]);
      acc[1][j] = wmma_bf16(a1, b, acc[1][j]);
    }
  }
  // C/D layout: VGPR r, lanes 0-15 -> (M=r, N=lane); lanes 16-31 -> (M=8+r, N=lane-16)
  const int mr = (lane >> 4) << 3;
  const int nc = lane & 15;
#pragma unroll
  for (int i = 0; i < 2; ++i) {
#pragma unroll
    for (int r = 0; r < 8; ++r) {
      const size_t row = (size_t)(m0 + 16 * i + mr + r) * N;
#pragma unroll
      for (int j = 0; j < 4; ++j)
        C[row + n0 + 16 * j + nc] = (OutT)acc[i][j][r];
    }
  }
}

// ---------------- flash attention (causal), bf16 WMMA ----------------
// grid: (B*H, T/128); block 256 = 8 waves; each wave owns 16 query rows.
__global__ __launch_bounds__(256) void attn_kernel(
    const __bf16* __restrict__ qb,   // (B*T, C) row-major, head h at col h*64
    const __bf16* __restrict__ kvb,  // (B*S, 2C) row-major, k head h at col h*64
    const __bf16* __restrict__ vT,   // (B,H,D,S)
    __bf16* __restrict__ Ob) {       // (B*T, C) row-major
  const int bh = blockIdx.x;
  const int b = bh >> 4, h = bh & 15;
  const int wid   = threadIdx.x >> 5;
  const int lane  = threadIdx.x & 31;
  const int mhalf = (lane >> 4) << 3; // 0 or 8
  const int ncol  = lane & 15;
  const int t0 = (blockIdx.y * 8 + wid) * 16;
  const float scale = 0.125f; // 1/sqrt(64)

  const __bf16* qbase = qb  + ((size_t)(b * TT + t0)) * CC + h * DD;
  const __bf16* kbase = kvb + (size_t)b * SSZ * (2 * CC) + h * DD;
  const __bf16* vbase = vT  + (size_t)bh * DD * SSZ;

  const bf16x16 q0 = load_frag_a(qbase, CC);
  const bf16x16 q1 = load_frag_a(qbase + 32, CC);

  __shared__ __bf16 ptile[8][16 * 32]; // one 16x32 P tile per wave
  __bf16* myP = &ptile[wid][0];

  float m_run[8], l_run[8], alpha[8];
  f32x8 o0 = {}, o1 = {}, o2 = {}, o3 = {};
#pragma unroll
  for (int r = 0; r < 8; ++r) { m_run[r] = -1e30f; l_run[r] = 0.f; }

  const int send = t0 + 16; // causal bound (exclusive)
  for (int s0 = 0; s0 < send; s0 += 32) {
    if (s0 + 32 < send) { // prefetch next chunk's K rows and V columns
      prefetch_frag(kbase + (size_t)(s0 + 32) * (2 * CC), 2 * CC);
      prefetch_frag(kbase + (size_t)(s0 + 48) * (2 * CC), 2 * CC);
#pragma unroll
      for (int jd = 0; jd < 4; ++jd)
        prefetch_frag(vbase + (size_t)(16 * jd) * SSZ + s0 + 32, SSZ);
    }
    // scores: two 16x16 tiles covering S columns [s0, s0+32)
    f32x8 sc0 = {}, sc1 = {};
    {
      const __bf16* kb0 = kbase + (size_t)s0 * (2 * CC);
      const __bf16* kb1 = kbase + (size_t)(s0 + 16) * (2 * CC);
      sc0 = wmma_bf16(q0, load_frag_b(kb0, 2 * CC), sc0);
      sc0 = wmma_bf16(q1, load_frag_b(kb0 + 32, 2 * CC), sc0);
      sc1 = wmma_bf16(q0, load_frag_b(kb1, 2 * CC), sc1);
      sc1 = wmma_bf16(q1, load_frag_b(kb1 + 32, 2 * CC), sc1);
    }
    // online softmax per row (rows split: lanes 0-15 hold M=r, lanes 16-31 M=8+r)
#pragma unroll
    for (int r = 0; r < 8; ++r) {
      const int trow = t0 + mhalf + r;
      float v0 = sc0[r] * scale; if (s0 + ncol > trow)      v0 = -1e30f;
      float v1 = sc1[r] * scale; if (s0 + 16 + ncol > trow) v1 = -1e30f;
      float mx = fmaxf(v0, v1);
#pragma unroll
      for (int off = 1; off < 16; off <<= 1) mx = fmaxf(mx, __shfl_xor(mx, off, 16));
      const float mn = fmaxf(m_run[r], mx);
      const float al = __expf(m_run[r] - mn);
      const float p0 = __expf(v0 - mn);
      const float p1 = __expf(v1 - mn);
      float rs = p0 + p1;
#pragma unroll
      for (int off = 1; off < 16; off <<= 1) rs += __shfl_xor(rs, off, 16);
      m_run[r] = mn;
      l_run[r] = l_run[r] * al + rs;
      alpha[r] = al;
      // stash P (bf16) row-major 16x32 in this wave's private LDS tile
      myP[(mhalf + r) * 32 + ncol]      = (__bf16)p0;
      myP[(mhalf + r) * 32 + 16 + ncol] = (__bf16)p1;
    }
    // re-stripe P into an A fragment via LDS (in-order DS within a wave)
    const bf16x16 pf = load_frag_a(myP, 32);
#pragma unroll
    for (int r = 0; r < 8; ++r) {
      o0[r] *= alpha[r]; o1[r] *= alpha[r]; o2[r] *= alpha[r]; o3[r] *= alpha[r];
    }
    o0 = wmma_bf16(pf, load_frag_b(vbase + (size_t)0  * SSZ + s0, SSZ), o0);
    o1 = wmma_bf16(pf, load_frag_b(vbase + (size_t)16 * SSZ + s0, SSZ), o1);
    o2 = wmma_bf16(pf, load_frag_b(vbase + (size_t)32 * SSZ + s0, SSZ), o2);
    o3 = wmma_bf16(pf, load_frag_b(vbase + (size_t)48 * SSZ + s0, SSZ), o3);
  }

  __bf16* obase = Ob + ((size_t)(b * TT + t0)) * CC + h * DD;
#pragma unroll
  for (int r = 0; r < 8; ++r) {
    const float inv = 1.0f / l_run[r];
    const size_t row = (size_t)(mhalf + r) * CC;
    obase[row + 0  + ncol] = (__bf16)(o0[r] * inv);
    obase[row + 16 + ncol] = (__bf16)(o1[r] * inv);
    obase[row + 32 + ncol] = (__bf16)(o2[r] * inv);
    obase[row + 48 + ncol] = (__bf16)(o3[r] * inv);
  }
}

// ---------------- host launcher ----------------

extern "C" void kernel_launch(void* const* d_in, const int* in_sizes, int n_in,
                              void* d_out, int out_size, void* d_ws, size_t ws_size,
                              hipStream_t stream) {
  const float* x   = (const float*)d_in[0]; // (B,T,C)
  const float* ctx = (const float*)d_in[1]; // (B,S,C)
  const float* Wq  = (const float*)d_in[2]; // (C,C)
  const float* Wkv = (const float*)d_in[3]; // (C,2C)
  const float* Wo  = (const float*)d_in[4]; // (C,C)
  float* out = (float*)d_out;               // (B,T,C) f32

  char* ws = (char*)d_ws;
  size_t off = 0;
  auto take = [&](size_t bytes) {
    char* p = ws + off;
    off += (bytes + 255) & ~(size_t)255;
    return p;
  };
  __bf16* xb   = (__bf16*)take((size_t)BQ * TT * CC * 2);
  __bf16* cb   = (__bf16*)take((size_t)BQ * SSZ * CC * 2);
  __bf16* WqT  = (__bf16*)take((size_t)CC * CC * 2);
  __bf16* WkvT = (__bf16*)take((size_t)2 * CC * CC * 2);
  __bf16* WoT  = (__bf16*)take((size_t)CC * CC * 2);
  __bf16* qb   = (__bf16*)take((size_t)BQ * TT * CC * 2);
  __bf16* kvb  = (__bf16*)take((size_t)BQ * SSZ * 2 * CC * 2);
  __bf16* vTb  = (__bf16*)take((size_t)BQ * HH * DD * SSZ * 2);
  __bf16* Ob   = (__bf16*)take((size_t)BQ * TT * CC * 2);
  (void)ws_size; (void)in_sizes; (void)n_in; (void)out_size;

  const size_t nX = (size_t)BQ * TT * CC; // 8388608
  cvt_f32_bf16_kernel<<<(unsigned)(nX / 256), 256, 0, stream>>>(x, xb, nX);
  cvt_f32_bf16_kernel<<<(unsigned)(nX / 256), 256, 0, stream>>>(ctx, cb, nX);

  transpose_f32_bf16_kernel<<<(CC * CC) / 256, 256, 0, stream>>>(Wq, WqT, CC, CC);
  transpose_f32_bf16_kernel<<<(CC * 2 * CC) / 256, 256, 0, stream>>>(Wkv, WkvT, CC, 2 * CC);
  transpose_f32_bf16_kernel<<<(CC * CC) / 256, 256, 0, stream>>>(Wo, WoT, CC, CC);

  // Q = xb * WqT^T : (8192 x 1024)
  gemm_bf16_kernel<__bf16><<<dim3(CC / 256, (BQ * TT) / 64), 256, 0, stream>>>(
      xb, WqT, qb, BQ * TT, CC, CC);
  // KV = cb * WkvT^T : (8192 x 2048)
  gemm_bf16_kernel<__bf16><<<dim3((2 * CC) / 256, (BQ * SSZ) / 64), 256, 0, stream>>>(
      cb, WkvT, kvb, BQ * SSZ, 2 * CC, CC);

  build_vT_kernel<<<(unsigned)(((size_t)BQ * HH * DD * SSZ) / 256), 256, 0, stream>>>(kvb, vTb);

  attn_kernel<<<dim3(BQ * HH, TT / 128), 256, 0, stream>>>(qb, kvb, vTb, Ob);

  // out = Ob * WoT^T : (8192 x 1024), fp32 output
  gemm_bf16_kernel<float><<<dim3(CC / 256, (BQ * TT) / 64), 256, 0, stream>>>(
      Ob, WoT, out, BQ * TT, CC, CC);
}
